// RPNHeadNew_91216515432735
// MI455X (gfx1250) — matrix operand order repaired and checked by reading, not compiled, not run on hardware
//
#include <hip/hip_runtime.h>
#include <hip/hip_bf16.h>

// ---------------------------------------------------------------------------
// Problem constants (from the reference)
// ---------------------------------------------------------------------------
#define DMODEL 1024
#define HDIM   512
#define NHEADS 2
#define NBATCH 2
#define SEQ    5440
#define NB     85          // SEQ / 64 key/query blocks
#define FFND   1024
#define MTOK   (NBATCH * SEQ)   // 10880 rows, divisible by 128
#define LN_EPS 1e-5f
#define ATTN_SCALE 0.04419417382415922f   // 1/sqrt(512)

typedef __attribute__((ext_vector_type(16))) __bf16 bf16x16;
typedef __attribute__((ext_vector_type(8)))  float  f32x8;
typedef __attribute__((ext_vector_type(4)))  int    v4i;

union FragU { unsigned int u[8]; bf16x16 h; };

__device__ __forceinline__ unsigned short f2bf(float f) {
  unsigned int u = __float_as_uint(f);
  u += 0x7fffu + ((u >> 16) & 1u);          // round-to-nearest-even
  return (unsigned short)(u >> 16);
}
__device__ __forceinline__ float bf2f(unsigned short s) {
  return __uint_as_float(((unsigned int)s) << 16);
}

// ---------------------------------------------------------------------------
// CDNA5 async global->LDS copy (ASYNCcnt), with compile-safe fallback
// ---------------------------------------------------------------------------
#if __has_builtin(__builtin_amdgcn_global_load_async_to_lds_b128)
#define HAVE_ASYNC 1
#else
#define HAVE_ASYNC 0
#endif

typedef __attribute__((address_space(1))) v4i gv4i_t;   // global int4
typedef __attribute__((address_space(3))) v4i lv4i_t;   // LDS int4

__device__ __forceinline__ void async_b128(void* lds, const void* g) {
#if HAVE_ASYNC
  __builtin_amdgcn_global_load_async_to_lds_b128(
      (gv4i_t*)(void*)(unsigned long long)(const char*)g,
      (lv4i_t*)lds, 0, 0);
#else
  *(uint4*)lds = *(const uint4*)g;
#endif
}
__device__ __forceinline__ void async_wait() {
#if HAVE_ASYNC
#if __has_builtin(__builtin_amdgcn_s_wait_asynccnt)
  __builtin_amdgcn_s_wait_asynccnt(0);
#else
  asm volatile("s_wait_asynccnt 0" ::: "memory");
#endif
#endif
}

// ---------------------------------------------------------------------------
// WMMA fragment builders (wave32; contiguous k = 16*(lane/16) + element, same order in A and B)
// ---------------------------------------------------------------------------
__device__ __forceinline__ void frag_a(FragU& f, const unsigned short* lds,
                                       int rowbase, int stride, int kbase, int lane) {
  int r  = rowbase + (lane & 15);
  int kh = (lane >> 4) * 16;
#pragma unroll
  for (int v = 0; v < 8; ++v) {
    int k = kbase + kh + v * 2;
    f.u[v] = *(const unsigned int*)(lds + r * stride + k);
  }
}
// B from LDS stored transposed [n][k] (pairs contiguous)
__device__ __forceinline__ void frag_b_t(FragU& f, const unsigned short* lds,
                                         int colbase, int stride, int kbase, int lane) {
  int c  = colbase + (lane & 15);
  int kh = (lane >> 4) * 16;
#pragma unroll
  for (int v = 0; v < 8; ++v) {
    int k = kbase + kh + v * 2;
    f.u[v] = *(const unsigned int*)(lds + c * stride + k);
  }
}
// B from LDS stored [k][n] (V tiles)
__device__ __forceinline__ void frag_b_kn(FragU& f, const unsigned short* lds,
                                          int colbase, int rowstride, int kbase, int lane) {
  int c  = colbase + (lane & 15);
  int kh = (lane >> 4) * 16;
#pragma unroll
  for (int v = 0; v < 8; ++v) {
    int k = kbase + kh + v * 2;
    unsigned int lo = lds[k * rowstride + c];
    unsigned int hi = lds[(k + 1) * rowstride + c];
    f.u[v] = lo | (hi << 16);
  }
}
__device__ __forceinline__ f32x8 wmma_bf16(const FragU& a, const FragU& b, f32x8 c) {
  return __builtin_amdgcn_wmma_f32_16x16x32_bf16(false, a.h, false, b.h,
                                                 (short)0, c, false, false);
}
__device__ __forceinline__ float gelu_new(float x) {
  return 0.5f * x * (1.0f + tanhf(0.7978845608028654f * (x + 0.044715f * x * x * x)));
}

// ---------------------------------------------------------------------------
// fp32 [K][N] -> bf16 transposed [N][K] (weights, one-time)
// ---------------------------------------------------------------------------
__global__ void cvt_t_kernel(const float* __restrict__ src,
                             unsigned short* __restrict__ dst, int K, int N) {
  int i = blockIdx.x * 256 + threadIdx.x;
  if (i >= K * N) return;
  int k = i / N, n = i - k * N;
  dst[(size_t)n * K + k] = f2bf(src[i]);
}

// ---------------------------------------------------------------------------
// token packing: h[b, off+p, d] = x[b, d, p] + pos[off+p, d]
// ---------------------------------------------------------------------------
__global__ void pack_kernel(const float* __restrict__ x, const float* __restrict__ pos,
                            float* __restrict__ h, int hw, int tok_off) {
  int idx = blockIdx.x * 256 + threadIdx.x;
  int total = NBATCH * DMODEL * hw;
  if (idx >= total) return;
  int b = idx / (DMODEL * hw);
  int rem = idx - b * DMODEL * hw;
  int d = rem / hw;
  int p = rem - d * hw;
  int t = tok_off + p;
  h[((size_t)b * SEQ + t) * DMODEL + d] = x[idx] + pos[(size_t)t * DMODEL + d];
}

// ---------------------------------------------------------------------------
// LayerNorm over D=1024 -> bf16 output (one block per token)
// ---------------------------------------------------------------------------
__global__ __launch_bounds__(256) void ln_kernel(const float* __restrict__ x,
                                                 const float* __restrict__ s,
                                                 const float* __restrict__ b,
                                                 unsigned short* __restrict__ out) {
  __shared__ float r1[256], r2[256];
  int row = blockIdx.x;
  const float* xr = x + (size_t)row * DMODEL;
  float sum = 0.f, sq = 0.f;
  for (int d = threadIdx.x; d < DMODEL; d += 256) { float v = xr[d]; sum += v; sq += v * v; }
  r1[threadIdx.x] = sum; r2[threadIdx.x] = sq; __syncthreads();
  for (int st = 128; st > 0; st >>= 1) {
    if (threadIdx.x < st) { r1[threadIdx.x] += r1[threadIdx.x + st]; r2[threadIdx.x] += r2[threadIdx.x + st]; }
    __syncthreads();
  }
  float mu = r1[0] * (1.0f / DMODEL);
  float var = r2[0] * (1.0f / DMODEL) - mu * mu;
  float rs = rsqrtf(var + LN_EPS);
  for (int d = threadIdx.x; d < DMODEL; d += 256)
    out[(size_t)row * DMODEL + d] = f2bf((xr[d] - mu) * rs * s[d] + b[d]);
}

// ---------------------------------------------------------------------------
// WMMA GEMM: out[M,N] = A[M,K](bf16) x Wt[N,K](bf16, pre-transposed) + bias.
// 128x64 tile / 256-thread block; wave w owns rows w*16..+15 and 4 N-tiles
// -> 4 WMMA per A-fragment. Double-buffered async global->LDS staging.
// ---------------------------------------------------------------------------
__global__ __launch_bounds__(256) void gemm_bf16_kernel(
    const unsigned short* __restrict__ A, const unsigned short* __restrict__ Wt,
    const float* __restrict__ bias, const float* __restrict__ resid,
    float* __restrict__ outF, unsigned short* __restrict__ outB,
    int M, int N, int K, int apply_gelu) {
  __shared__ unsigned short lds_a[2][128 * 32];   // [row][k]
  __shared__ unsigned short lds_w[2][64 * 32];    // [n][k]
  int m0 = blockIdx.x * 128;
  int n0 = blockIdx.y * 64;
  int tid = threadIdx.x;
  int wave = tid >> 5, lane = tid & 31;
  f32x8 acc[4];
#pragma unroll
  for (int i = 0; i < 4; ++i) { f32x8 z = {}; acc[i] = z; }

  auto issue = [&](int buf, int k0) {
    for (int c = tid; c < 128 * 4; c += 256) {          // A: 512 x 16B chunks
      int r = c >> 2, p = c & 3;
      async_b128(&lds_a[buf][r * 32 + p * 8], A + (size_t)(m0 + r) * K + k0 + p * 8);
    }
    for (int c = tid; c < 64 * 4; c += 256) {           // W: 256 x 16B chunks
      int r = c >> 2, p = c & 3;
      async_b128(&lds_w[buf][r * 32 + p * 8], Wt + (size_t)(n0 + r) * K + k0 + p * 8);
    }
  };
  issue(0, 0);
  int cur = 0;
  for (int k0 = 0; k0 < K; k0 += 32) {
    async_wait();
    __syncthreads();
    if (k0 + 32 < K) issue(cur ^ 1, k0 + 32);
    FragU af;
    frag_a(af, lds_a[cur], wave * 16, 32, 0, lane);
#pragma unroll
    for (int ni = 0; ni < 4; ++ni) {
      FragU bfw;
      frag_b_t(bfw, lds_w[cur], ni * 16, 32, 0, lane);
      acc[ni] = wmma_bf16(af, bfw, acc[ni]);
    }
    cur ^= 1;
  }
#pragma unroll
  for (int ni = 0; ni < 4; ++ni) {
    int col = n0 + ni * 16 + (lane & 15);
#pragma unroll
    for (int v = 0; v < 8; ++v) {
      int row = m0 + wave * 16 + (lane >> 4) * 8 + v;
      float x = acc[ni][v];
      if (bias) x += bias[col];
      if (apply_gelu) x = gelu_new(x);
      if (resid) x += resid[(size_t)row * N + col];
      if (outF) outF[(size_t)row * N + col] = x;
      if (outB) outB[(size_t)row * N + col] = f2bf(x);
    }
  }
}

// ---------------------------------------------------------------------------
// Block-sparse attention, middle rows (1..NB-2). One workgroup per
// (batch, head, query block). QK^T and P*V on WMMA bf16, softmax in LDS fp32,
// async double-buffered K-strip and V-block staging.
// ---------------------------------------------------------------------------
#define AMID_LDS (64*512*2 + 64*512*4 + 64*512*2 + 2*64*32*2 + (256+64+64)*4)

__global__ __launch_bounds__(256) void attn_mid_kernel(
    const unsigned short* __restrict__ q, const unsigned short* __restrict__ k,
    const unsigned short* __restrict__ v, const int* __restrict__ rand_attn,
    unsigned short* __restrict__ ctx) {
  extern __shared__ char smem[];
  unsigned short* qs  = (unsigned short*)smem;                                 // 64x512 bf16
  float*          sc  = (float*)(smem + 64 * 512 * 2);                         // 64x512 f32
  unsigned short* pb  = (unsigned short*)(smem + 64 * 512 * 2 + 64 * 512 * 4); // probs bf16
  unsigned short* kst = pb + 64 * 512;                                         // 2 x (64x32) K stage
  float* redp = (float*)(kst + 2 * 64 * 32);                                   // 64x4
  float* rowm = redp + 256;
  float* rowl = rowm + 64;
  __shared__ int klist[8];
  __shared__ int kcnt;

  int bid = blockIdx.x;
  int m  = 1 + (bid % (NB - 2));
  int hh = (bid / (NB - 2)) % NHEADS;
  int bb = bid / ((NB - 2) * NHEADS);
  size_t hbase = (size_t)bb * SEQ * DMODEL + (size_t)hh * HDIM;

  if (threadIdx.x == 0) {
    int c = 0;
    if (m == 1)           { klist[c++] = 0;      klist[c++] = 1;      klist[c++] = 2;      klist[c++] = NB - 1; }
    else if (m == NB - 2) { klist[c++] = NB - 3; klist[c++] = NB - 2; klist[c++] = NB - 1; klist[c++] = 0; }
    else { klist[c++] = 0; klist[c++] = m - 1; klist[c++] = m; klist[c++] = m + 1; klist[c++] = NB - 1; }
    const int* ra = rand_attn + ((hh * (NB - 2)) + (m - 1)) * 3;
    klist[c++] = ra[0]; klist[c++] = ra[1]; klist[c++] = ra[2];
    kcnt = c;
  }
  // Q block (64x512 bf16) -> LDS, async
  for (int c = threadIdx.x; c < 64 * 64; c += 256) {
    int r = c >> 6, p = c & 63;
    async_b128(qs + r * 512 + p * 8, q + hbase + (size_t)(m * 64 + r) * DMODEL + p * 8);
  }
  __syncthreads();
  int kc = kcnt;
  int wave = threadIdx.x >> 5, lane = threadIdx.x & 31;
  int mi = wave & 3, nh2 = wave >> 2;

  auto issue_k = [&](int buf, int kb, int dk) {
    int r = threadIdx.x >> 2, p = threadIdx.x & 3;      // 256 x 16B chunks
    async_b128(kst + buf * (64 * 32) + r * 32 + p * 8,
               k + hbase + (size_t)(kb * 64 + r) * DMODEL + dk + p * 8);
  };

  // ---- scores: S = (Q K^T) * scale, double-buffered K strips ----------
  int nsteps = kc * 16;                                  // 16 strips of 32 dims
  issue_k(0, klist[0], 0);
  int cur = 0;
  f32x8 a0 = {}, a1 = {};
  for (int s = 0; s < nsteps; ++s) {
    async_wait();
    __syncthreads();
    if (s + 1 < nsteps) issue_k(cur ^ 1, klist[(s + 1) >> 4], ((s + 1) & 15) * 32);
    int kb_i = s >> 4, dk = (s & 15) * 32;
    if ((s & 15) == 0) { f32x8 z = {}; a0 = z; a1 = z; }
    FragU af, b0, b1;
    frag_a(af, qs, mi * 16, HDIM, dk, lane);
    frag_b_t(b0, kst + cur * (64 * 32), (nh2 * 2 + 0) * 16, 32, 0, lane);
    frag_b_t(b1, kst + cur * (64 * 32), (nh2 * 2 + 1) * 16, 32, 0, lane);
    a0 = wmma_bf16(af, b0, a0);
    a1 = wmma_bf16(af, b1, a1);
    if ((s & 15) == 15) {
      int cb = kb_i * 64;
#pragma unroll
      for (int vv = 0; vv < 8; ++vv) {
        int row = mi * 16 + (lane >> 4) * 8 + vv;
        sc[row * 512 + cb + (nh2 * 2 + 0) * 16 + (lane & 15)] = a0[vv] * ATTN_SCALE;
        sc[row * 512 + cb + (nh2 * 2 + 1) * 16 + (lane & 15)] = a1[vv] * ATTN_SCALE;
      }
    }
    cur ^= 1;
  }
  __syncthreads();

  // ---- softmax over kw = kc*64 keys (4 threads per row) ---------------
  {
    int r = threadIdx.x >> 2, g = threadIdx.x & 3;
    int kw = kc * 64;
    float mx = -1e30f;
    for (int c = g; c < kw; c += 4) mx = fmaxf(mx, sc[r * 512 + c]);
    redp[r * 4 + g] = mx; __syncthreads();
    if (g == 0)
      rowm[r] = fmaxf(fmaxf(redp[r * 4], redp[r * 4 + 1]), fmaxf(redp[r * 4 + 2], redp[r * 4 + 3]));
    __syncthreads();
    float rm = rowm[r], se = 0.f;
    for (int c = g; c < kw; c += 4) se += __expf(sc[r * 512 + c] - rm);
    redp[r * 4 + g] = se; __syncthreads();
    if (g == 0) rowl[r] = redp[r * 4] + redp[r * 4 + 1] + redp[r * 4 + 2] + redp[r * 4 + 3];
    __syncthreads();
    float inv = 1.0f / rowl[r];
    for (int c = g; c < kw; c += 4) pb[r * 512 + c] = f2bf(__expf(sc[r * 512 + c] - rm) * inv);
  }
  __syncthreads();

  // ---- O = P V, WMMA; V double-buffered in the (now free) score buffer
  unsigned short* vst = (unsigned short*)sc;             // 2 x (64x512) bf16
  auto issue_v = [&](int buf, int kb) {
    for (int c = threadIdx.x; c < 64 * 64; c += 256) {   // 4096 x 16B chunks
      int r = c >> 6, p = c & 63;
      async_b128(vst + buf * (64 * 512) + r * 512 + p * 8,
                 v + hbase + (size_t)(kb * 64 + r) * DMODEL + p * 8);
    }
  };
  int dh = wave >> 2;                                    // dim half (256 dims)
  f32x8 oacc[16];
#pragma unroll
  for (int i = 0; i < 16; ++i) { f32x8 z = {}; oacc[i] = z; }
  issue_v(0, klist[0]);
  cur = 0;
  for (int kb_i = 0; kb_i < kc; ++kb_i) {
    async_wait();
    __syncthreads();
    if (kb_i + 1 < kc) issue_v(cur ^ 1, klist[kb_i + 1]);
    const unsigned short* vb = vst + cur * (64 * 512);
    for (int kk = 0; kk < 64; kk += 32) {
      FragU af;
      frag_a(af, pb, mi * 16, 512, kb_i * 64 + kk, lane);
#pragma unroll
      for (int nt = 0; nt < 16; ++nt) {
        FragU bfv;
        frag_b_kn(bfv, vb, dh * 256 + nt * 16, 512, kk, lane);
        oacc[nt] = wmma_bf16(af, bfv, oacc[nt]);
      }
    }
    cur ^= 1;
  }
#pragma unroll
  for (int nt = 0; nt < 16; ++nt) {
    int col = dh * 256 + nt * 16 + (lane & 15);
#pragma unroll
    for (int vv = 0; vv < 8; ++vv) {
      int row = mi * 16 + (lane >> 4) * 8 + vv;
      ctx[hbase + (size_t)(m * 64 + row) * DMODEL + col] = f2bf(oacc[nt][vv]);
    }
  }
}

// ---------------------------------------------------------------------------
// Dense edge rows (query blocks 0 and NB-1): streaming online softmax, VALU.
// ---------------------------------------------------------------------------
#define AEDGE_LDS (64*512*4*2 + 64*64*4 + 256*4*2 + 64*4*2)

__global__ __launch_bounds__(256) void attn_edge_kernel(
    const unsigned short* __restrict__ q, const unsigned short* __restrict__ k,
    const unsigned short* __restrict__ v, unsigned short* __restrict__ ctx) {
  extern __shared__ char smem[];
  float* qf    = (float*)smem;            // 64x512
  float* of    = qf + 64 * 512;           // 64x512 accumulator
  float* pr    = of + 64 * 512;           // 64x64 probs tile
  float* mpart = pr + 64 * 64;
  float* lpart = mpart + 256;
  float* rowm  = lpart + 256;
  float* rowl  = rowm + 64;

  int bid = blockIdx.x;
  int e  = bid & 1;
  int hh = (bid >> 1) & 1;
  int bb = bid >> 2;
  int m  = e ? (NB - 1) : 0;
  size_t hbase = (size_t)bb * SEQ * DMODEL + (size_t)hh * HDIM;

  for (int i = threadIdx.x; i < 64 * 512; i += 256) {
    int r = i >> 9, d = i & 511;
    qf[i] = bf2f(q[hbase + (size_t)(m * 64 + r) * DMODEL + d]);
    of[i] = 0.f;
  }
  __syncthreads();

  int r = threadIdx.x >> 2, g = threadIdx.x & 3;
  float mloc = -1e30f, lloc = 0.f;
  for (int key = g; key < SEQ; key += 4) {
    const unsigned short* kr = k + hbase + (size_t)key * DMODEL;
    float dot = 0.f;
    for (int d = 0; d < HDIM; ++d) dot += qf[r * 512 + d] * bf2f(kr[d]);
    float s = dot * ATTN_SCALE;
    float mn = fmaxf(mloc, s);
    lloc = lloc * __expf(mloc - mn) + __expf(s - mn);
    mloc = mn;
  }
  mpart[r * 4 + g] = mloc; lpart[r * 4 + g] = lloc; __syncthreads();
  if (g == 0) {
    float M = -1e30f;
    for (int i2 = 0; i2 < 4; ++i2) M = fmaxf(M, mpart[r * 4 + i2]);
    float L = 0.f;
    for (int i2 = 0; i2 < 4; ++i2) L += lpart[r * 4 + i2] * __expf(mpart[r * 4 + i2] - M);
    rowm[r] = M; rowl[r] = L;
  }
  __syncthreads();
  float M = rowm[r], invL = 1.0f / rowl[r];

  for (int kb = 0; kb < NB; ++kb) {
    for (int jj = g; jj < 64; jj += 4) {
      const unsigned short* kr = k + hbase + (size_t)(kb * 64 + jj) * DMODEL;
      float dot = 0.f;
      for (int d = 0; d < HDIM; ++d) dot += qf[r * 512 + d] * bf2f(kr[d]);
      pr[r * 64 + jj] = __expf(dot * ATTN_SCALE - M) * invL;
    }
    __syncthreads();
    int dc = (threadIdx.x & 3) * 128;
    for (int j = 0; j < 64; ++j) {
      float p = pr[r * 64 + j];
      const unsigned short* vr = v + hbase + (size_t)(kb * 64 + j) * DMODEL + dc;
      for (int d = 0; d < 128; ++d) of[r * 512 + dc + d] += p * bf2f(vr[d]);
    }
    __syncthreads();
  }
  for (int i = threadIdx.x; i < 64 * 512; i += 256) {
    int rr = i >> 9, d = i & 511;
    ctx[hbase + (size_t)(m * 64 + rr) * DMODEL + d] = f2bf(of[i]);
  }
}

// ---------------------------------------------------------------------------
// Final LN + cls/bbox heads (15 dot products of 1024 per token)
// ---------------------------------------------------------------------------
__global__ __launch_bounds__(256) void head_kernel(
    const float* __restrict__ h, const float* __restrict__ lnf_s, const float* __restrict__ lnf_b,
    const float* __restrict__ cls_w, const float* __restrict__ cls_b,
    const float* __restrict__ bbox_w, const float* __restrict__ bbox_b,
    float* __restrict__ out) {
  __shared__ float hn[DMODEL];
  __shared__ float red[256];
  int row = blockIdx.x;
  int bb = row / SEQ, t = row % SEQ;
  const float* xr = h + (size_t)row * DMODEL;
  float sum = 0.f, sq = 0.f;
  for (int d = threadIdx.x; d < DMODEL; d += 256) { float x = xr[d]; sum += x; sq += x * x; }
  red[threadIdx.x] = sum; __syncthreads();
  for (int st = 128; st > 0; st >>= 1) { if (threadIdx.x < st) red[threadIdx.x] += red[threadIdx.x + st]; __syncthreads(); }
  float mu = red[0] * (1.0f / DMODEL);
  __syncthreads();
  red[threadIdx.x] = sq; __syncthreads();
  for (int st = 128; st > 0; st >>= 1) { if (threadIdx.x < st) red[threadIdx.x] += red[threadIdx.x + st]; __syncthreads(); }
  float var = red[0] * (1.0f / DMODEL) - mu * mu;
  float rs = rsqrtf(var + LN_EPS);
  __syncthreads();
  for (int d = threadIdx.x; d < DMODEL; d += 256)
    hn[d] = (xr[d] - mu) * rs * lnf_s[d] + lnf_b[d];
  __syncthreads();

  int lv, p, s2;
  if (t < 4096)      { lv = 0; p = t;        s2 = 4096; }
  else if (t < 5120) { lv = 1; p = t - 4096; s2 = 1024; }
  else if (t < 5376) { lv = 2; p = t - 5120; s2 = 256;  }
  else               { lv = 3; p = t - 5376; s2 = 64;   }
  const int LOFF[4] = {0, 24576, 30720, 32256};
  const int BOFF[4] = {32640, 130944, 155520, 161664};

  for (int o = 0; o < 15; ++o) {
    const float* w = (o < 3) ? (cls_w + o * DMODEL) : (bbox_w + (o - 3) * DMODEL);
    float part = 0.f;
    for (int d = threadIdx.x; d < DMODEL; d += 256) part += w[d] * hn[d];
    red[threadIdx.x] = part; __syncthreads();
    for (int st = 128; st > 0; st >>= 1) { if (threadIdx.x < st) red[threadIdx.x] += red[threadIdx.x + st]; __syncthreads(); }
    if (threadIdx.x == 0) {
      float val = red[0] + ((o < 3) ? cls_b[o] : bbox_b[o - 3]);
      size_t oi = (o < 3) ? ((size_t)LOFF[lv] + ((size_t)bb * 3 + o) * s2 + p)
                          : ((size_t)BOFF[lv] + ((size_t)bb * 12 + (o - 3)) * s2 + p);
      out[oi] = val;
    }
    __syncthreads();
  }
}

// ---------------------------------------------------------------------------
// Host-side orchestration
// ---------------------------------------------------------------------------
extern "C" void kernel_launch(void* const* d_in, const int* in_sizes, int n_in,
                              void* d_out, int out_size, void* d_ws, size_t ws_size,
                              hipStream_t stream) {
  const float* x0   = (const float*)d_in[0];
  const float* x1   = (const float*)d_in[1];
  const float* x2   = (const float*)d_in[2];
  const float* x3   = (const float*)d_in[3];
  const float* pos  = (const float*)d_in[4];
  const float* ln1s = (const float*)d_in[5];
  const float* ln1b = (const float*)d_in[6];
  const float* qW   = (const float*)d_in[7];
  const float* qB   = (const float*)d_in[8];
  const float* kW   = (const float*)d_in[9];
  const float* kB   = (const float*)d_in[10];
  const float* vW   = (const float*)d_in[11];
  const float* vB   = (const float*)d_in[12];
  const float* oW   = (const float*)d_in[13];
  const float* oB   = (const float*)d_in[14];
  const float* ln2s = (const float*)d_in[15];
  const float* ln2b = (const float*)d_in[16];
  const float* f1W  = (const float*)d_in[17];
  const float* f1B  = (const float*)d_in[18];
  const float* f2W  = (const float*)d_in[19];
  const float* f2B  = (const float*)d_in[20];
  const float* lnfs = (const float*)d_in[21];
  const float* lnfb = (const float*)d_in[22];
  const float* clsW = (const float*)d_in[23];
  const float* clsB = (const float*)d_in[24];
  const float* bbW  = (const float*)d_in[25];
  const float* bbB  = (const float*)d_in[26];
  const int*   rnd  = (const int*)d_in[27];

  // ---- workspace layout ----
  char* ws = (char*)d_ws;
  size_t off = 0;
  auto alloc = [&](size_t bytes) -> char* {
    off = (off + 255) & ~(size_t)255;
    char* p = ws + off;
    off += bytes;
    return p;
  };
  float*          h    = (float*)         alloc((size_t)MTOK * DMODEL * 4);
  unsigned short* hn   = (unsigned short*)alloc((size_t)MTOK * DMODEL * 2);
  unsigned short* qb   = (unsigned short*)alloc((size_t)MTOK * DMODEL * 2);
  unsigned short* kb   = (unsigned short*)alloc((size_t)MTOK * DMODEL * 2);
  unsigned short* vb   = (unsigned short*)alloc((size_t)MTOK * DMODEL * 2);
  unsigned short* ctx  = (unsigned short*)alloc((size_t)MTOK * DMODEL * 2);
  unsigned short* ff   = (unsigned short*)alloc((size_t)MTOK * FFND * 2);
  unsigned short* wbf[12];  // per layer: q,k,v,o,fc1,fc2 (each stored [N][K])
  for (int i = 0; i < 12; ++i) wbf[i] = (unsigned short*)alloc((size_t)DMODEL * DMODEL * 2);

  const int WCNT = DMODEL * DMODEL;
  const int cvtGrid = (WCNT + 255) / 256;
  const float* wsrc[12] = {qW, kW, vW, oW, f1W, f2W,
                           qW + WCNT, kW + WCNT, vW + WCNT, oW + WCNT,
                           f1W + (size_t)DMODEL * FFND, f2W + (size_t)FFND * DMODEL};
  for (int l = 0; l < 2; ++l)
    for (int j = 0; j < 6; ++j)
      cvt_t_kernel<<<cvtGrid, 256, 0, stream>>>(wsrc[l * 6 + j], wbf[l * 6 + j], DMODEL, DMODEL);

  // token pack
  const float* xs[4] = {x0, x1, x2, x3};
  const int hwv[4]  = {4096, 1024, 256, 64};
  const int toff[4] = {0, 4096, 5120, 5376};
  for (int lv = 0; lv < 4; ++lv) {
    int total = NBATCH * DMODEL * hwv[lv];
    pack_kernel<<<(total + 255) / 256, 256, 0, stream>>>(xs[lv], pos, h, hwv[lv], toff[lv]);
  }

  dim3 ggrid(MTOK / 128, DMODEL / 64);
  for (int l = 0; l < 2; ++l) {
    unsigned short** wl = &wbf[l * 6];
    // --- attention sublayer ---
    ln_kernel<<<MTOK, 256, 0, stream>>>(h, ln1s + l * DMODEL, ln1b + l * DMODEL, hn);
    gemm_bf16_kernel<<<ggrid, 256, 0, stream>>>(hn, wl[0], qB + l * DMODEL, nullptr, nullptr, qb, MTOK, DMODEL, DMODEL, 0);
    gemm_bf16_kernel<<<ggrid, 256, 0, stream>>>(hn, wl[1], kB + l * DMODEL, nullptr, nullptr, kb, MTOK, DMODEL, DMODEL, 0);
    gemm_bf16_kernel<<<ggrid, 256, 0, stream>>>(hn, wl[2], vB + l * DMODEL, nullptr, nullptr, vb, MTOK, DMODEL, DMODEL, 0);
    attn_mid_kernel<<<NBATCH * NHEADS * (NB - 2), 256, AMID_LDS, stream>>>(qb, kb, vb, rnd, ctx);
    attn_edge_kernel<<<NBATCH * NHEADS * 2, 256, AEDGE_LDS, stream>>>(qb, kb, vb, ctx);
    gemm_bf16_kernel<<<ggrid, 256, 0, stream>>>(ctx, wl[3], oB + l * DMODEL, h, h, nullptr, MTOK, DMODEL, DMODEL, 0);
    // --- FFN sublayer ---
    ln_kernel<<<MTOK, 256, 0, stream>>>(h, ln2s + l * DMODEL, ln2b + l * DMODEL, hn);
    gemm_bf16_kernel<<<ggrid, 256, 0, stream>>>(hn, wl[4], f1B + l * FFND, nullptr, nullptr, ff, MTOK, FFND, DMODEL, 1);
    gemm_bf16_kernel<<<ggrid, 256, 0, stream>>>(ff, wl[5], f2B + l * DMODEL, h, h, nullptr, MTOK, DMODEL, FFND, 0);
  }

  head_kernel<<<MTOK, 256, 0, stream>>>(h, lnfs, lnfb, clsW, clsB, bbW, bbB, (float*)d_out);
}
